// MyModel_19885698580986
// MI455X (gfx1250) — matrix operand order repaired and checked by reading, not compiled, not run on hardware
//
#include <hip/hip_runtime.h>
#include <hip/hip_bf16.h>

// ---------------------------------------------------------------------------
// GCN (two GCNConv branches + mean pool + linear head) for MI455X / gfx1250.
// GEMM: v_wmma_f32_16x16x32_bf16, fp32->bf16 via vector fptrunc (RNE; lowers
// to v_cvt_pk_bf16_f32 when the backend has the native pattern).
// W staged in LDS per 256-K panel, layout [col][k-pair] (+4 pad) so each B
// fragment is two contiguous ds_load_b128 with zero register shuffling and
// minimal bank conflicts. Wave tile = 32(M) x 64(N) => 8 WMMA / 32-K step,
// inner loop barrier- and branch-free. Scatter: native f32 global atomics
// resolving in the 192MB L2 (accumulators are 51/102MB).
// ---------------------------------------------------------------------------

typedef __attribute__((ext_vector_type(16))) __bf16       v16bf;
typedef __attribute__((ext_vector_type(2)))  __bf16       v2bf;
typedef __attribute__((ext_vector_type(2)))  float        v2f;
typedef __attribute__((ext_vector_type(8)))  float        v8f;
typedef __attribute__((ext_vector_type(8)))  unsigned int v8u;

#define NUM_GRAPHS 128
#define PANEL 256      // K-values staged per LDS panel
#define LDST 132       // LDS row stride in dwords: 128 k-pairs + 4 pad

__device__ __forceinline__ unsigned int bf16pk(float lo, float hi) {
  // vector fptrunc -> v_cvt_pk_bf16_f32 on targets with native bf16 cvt
  v2f t;
  t.x = lo;
  t.y = hi;
  v2bf r = __builtin_convertvector(t, v2bf);  // lo->bits[15:0], hi->[31:16]
  return __builtin_bit_cast(unsigned int, r);
}

// --------------------------- small helper kernels ---------------------------

__global__ void gcn_fill1_kernel(float* __restrict__ p, int n) {
  int i = blockIdx.x * blockDim.x + threadIdx.x;
  if (i < n) p[i] = 1.0f;
}

__global__ void gcn_edge_deg_kernel(const int* __restrict__ ei, int E,
                                    float* __restrict__ deg) {
  int e = blockIdx.x * blockDim.x + threadIdx.x;
  if (e < E) unsafeAtomicAdd(&deg[ei[E + e]], 1.0f);  // dst row of edge_index
}

__global__ void gcn_rsqrt_kernel(float* __restrict__ p, int n) {
  int i = blockIdx.x * blockDim.x + threadIdx.x;
  if (i < n) p[i] = __frsqrt_rn(p[i]);  // deg >= 1 always (self loops)
}

// ----------------------------- A fragment load ------------------------------
// 16-bit A layout (ISA 7.12.2): lane m = lane&15; half 0: K {0..7, 16..23},
// half 1: K {8..15, 24..31} relative to k-step base.

__device__ __forceinline__ v16bf load_a_frag(const float* __restrict__ row,
                                             int kA, int K, bool fast) {
  v8u au;
  if (fast) {
    float4 f0 = *(const float4*)(row + kA);
    float4 f1 = *(const float4*)(row + kA + 4);
    float4 f2 = *(const float4*)(row + kA + 16);
    float4 f3 = *(const float4*)(row + kA + 20);
    __builtin_prefetch(row + kA + 32, 0, 0);  // global_prefetch next k-step
    au[0] = bf16pk(f0.x, f0.y); au[1] = bf16pk(f0.z, f0.w);
    au[2] = bf16pk(f1.x, f1.y); au[3] = bf16pk(f1.z, f1.w);
    au[4] = bf16pk(f2.x, f2.y); au[5] = bf16pk(f2.z, f2.w);
    au[6] = bf16pk(f3.x, f3.y); au[7] = bf16pk(f3.z, f3.w);
  } else {  // K tail (r branch): guarded scalar loads, zero-pad
    float g[16];
#pragma unroll
    for (int e = 0; e < 8; ++e) {
      int k1 = kA + e;      g[e]     = (k1 < K) ? row[k1] : 0.0f;
      int k2 = kA + 16 + e; g[8 + e] = (k2 < K) ? row[k2] : 0.0f;
    }
#pragma unroll
    for (int j = 0; j < 8; ++j) au[j] = bf16pk(g[2 * j], g[2 * j + 1]);
  }
  return __builtin_bit_cast(v16bf, au);
}

// ------------------------------- WMMA GEMM ----------------------------------
// XW[M x 64] = X[M x K] @ W[K x 64].  One wave -> 32(M) x 64(N) output.
// Block = 256 threads = 8 waves = 256 rows.  W staged per 256-K panel in LDS
// as packed bf16 pairs, layout [col][k-pair] (stride LDST, padded).

__global__ __launch_bounds__(256) void gcn_gemm_wmma_kernel(
    const float* __restrict__ X, const float* __restrict__ W,
    float* __restrict__ XW, int M, int K) {
  __shared__ unsigned int ldsB[64 * LDST];  // ~33KB

  const int tid  = threadIdx.x;
  const int lane = tid & 31;
  const int wid  = tid >> 5;
  const int half = (lane >> 4) & 1;  // wave32 half
  const int lm   = lane & 15;

  int mBase = blockIdx.x * 256 + wid * 32;
  int m0 = mBase;
  int m1 = mBase + 16;
  if (m1 > M - 16) m1 = M - 16;  // duplicate tail work; deterministic
  if (m0 > M - 16) m0 = M - 16;

  const float* row0 = X + (size_t)(m0 + lm) * (size_t)K;
  const float* row1 = X + (size_t)(m1 + lm) * (size_t)K;

  v8f acc0[4] = {};
  v8f acc1[4] = {};

  for (int kp0 = 0; kp0 < K; kp0 += PANEL) {
    const int kpan = (K - kp0 < PANEL) ? (K - kp0) : PANEL;
    __syncthreads();  // protect previous panel reads
    if (kpan == PANEL) {
      // fast staging: thread -> one col, 4 k-pairs (8 W rows), one b128 store
#pragma unroll
      for (int q = 0; q < 8; ++q) {
        int qi  = tid + q * 256;        // 2048 quad entries
        int col = qi & 63;
        int kq  = qi >> 6;              // k-quad 0..31
        const float* wp = W + (size_t)(kp0 + 8 * kq) * 64 + col;
        uint4 pk4;
        pk4.x = bf16pk(wp[0],   wp[64]);
        pk4.y = bf16pk(wp[128], wp[192]);
        pk4.z = bf16pk(wp[256], wp[320]);
        pk4.w = bf16pk(wp[384], wp[448]);
        *(uint4*)&ldsB[col * LDST + 4 * kq] = pk4;
      }
    } else {
      // tail panel (r branch, K=120): guarded scalar staging, zero-pad
      for (int q = 0; q < 32; ++q) {
        int idx = tid + q * 256;        // 8192 dword entries
        int kp  = idx >> 6;
        int col = idx & 63;
        int ke  = kp0 + 2 * kp;
        float lo = (ke     < K) ? W[(size_t)ke * 64 + col]       : 0.0f;
        float hi = (ke + 1 < K) ? W[(size_t)(ke + 1) * 64 + col] : 0.0f;
        ldsB[col * LDST + kp] = bf16pk(lo, hi);
      }
    }
    __syncthreads();

    auto step = [&](int k0, bool fast) {
      const int kA = kp0 + k0 + half * 8;
      v16bf a0 = load_a_frag(row0, kA, K, fast);
      v16bf a1 = load_a_frag(row1, kA, K, fast);
      const int kb = (k0 >> 1) + half * 4;  // base k-pair for this half
#pragma unroll
      for (int nb = 0; nb < 4; ++nb) {
        const unsigned int* bp = &ldsB[(nb * 16 + lm) * LDST + kb];
        uint4 q0 = *(const uint4*)(bp);      // ds_load_b128
        uint4 q1 = *(const uint4*)(bp + 8);  // ds_load_b128
        v8u bu;
        bu[0] = q0.x; bu[1] = q0.y; bu[2] = q0.z; bu[3] = q0.w;
        bu[4] = q1.x; bu[5] = q1.y; bu[6] = q1.z; bu[7] = q1.w;
        v16bf b = __builtin_bit_cast(v16bf, bu);
        acc0[nb] = __builtin_amdgcn_wmma_f32_16x16x32_bf16(
            false, a0, false, b, (short)0, acc0[nb], false, false);
        acc1[nb] = __builtin_amdgcn_wmma_f32_16x16x32_bf16(
            false, a1, false, b, (short)0, acc1[nb], false, false);
      }
    };

    const int kfull = kpan & ~31;   // branch-free main steps
    for (int k0 = 0; k0 < kfull; k0 += 32) step(k0, true);
    if (kfull < kpan) step(kfull, false);  // at most one guarded tail step
  }

  // store D: VGPR v -> row m + half*8 + v, col nb*16 + lm
#pragma unroll
  for (int nb = 0; nb < 4; ++nb)
#pragma unroll
    for (int v = 0; v < 8; ++v) {
      XW[(size_t)(m0 + half * 8 + v) * 64 + nb * 16 + lm] = acc0[nb][v];
      XW[(size_t)(m1 + half * 8 + v) * 64 + nb * 16 + lm] = acc1[nb][v];
    }
}

// ------------------------------ edge scatter --------------------------------
// 16 threads per edge, float4 per thread; atomics land in L2.

__global__ __launch_bounds__(256) void gcn_edge_scatter_kernel(
    const int* __restrict__ ei, int E, const float* __restrict__ xw,
    const float* __restrict__ dinv, float* __restrict__ acc) {
  long long t = (long long)blockIdx.x * blockDim.x + threadIdx.x;
  int e = (int)(t >> 4);
  int f = (int)(t & 15) * 4;
  if (e >= E) return;
  int s = ei[e];
  int d = ei[E + e];
  float nrm = dinv[s] * dinv[d];
  float4 xv = *(const float4*)(xw + (size_t)s * 64 + f);
  float* dst = acc + (size_t)d * 64 + f;
  unsafeAtomicAdd(dst + 0, xv.x * nrm);
  unsafeAtomicAdd(dst + 1, xv.y * nrm);
  unsafeAtomicAdd(dst + 2, xv.z * nrm);
  unsafeAtomicAdd(dst + 3, xv.w * nrm);
}

// ---------------------- relu + self-loop + mean-pool ------------------------

__global__ __launch_bounds__(256) void gcn_finalize_kernel(
    const float* __restrict__ xw, const float* __restrict__ acc,
    const float* __restrict__ dinv, const float* __restrict__ bias,
    const int* __restrict__ batch, float* __restrict__ gsum,
    float* __restrict__ gcnt, int N) {
  long long t = (long long)blockIdx.x * blockDim.x + threadIdx.x;
  int i = (int)(t >> 4);
  int f = (int)(t & 15) * 4;
  if (i >= N) return;
  float di = dinv[i];
  float sl = di * di;
  float4 av = *(const float4*)(acc + (size_t)i * 64 + f);
  float4 xv = *(const float4*)(xw + (size_t)i * 64 + f);
  float4 bv = *(const float4*)(bias + f);
  float v0 = fmaxf(av.x + xv.x * sl + bv.x, 0.0f);
  float v1 = fmaxf(av.y + xv.y * sl + bv.y, 0.0f);
  float v2 = fmaxf(av.z + xv.z * sl + bv.z, 0.0f);
  float v3 = fmaxf(av.w + xv.w * sl + bv.w, 0.0f);
  int g = batch[i];
  float* dst = gsum + (size_t)g * 64 + f;
  unsafeAtomicAdd(dst + 0, v0);
  unsafeAtomicAdd(dst + 1, v1);
  unsafeAtomicAdd(dst + 2, v2);
  unsafeAtomicAdd(dst + 3, v3);
  if (f == 0) unsafeAtomicAdd(&gcnt[g], 1.0f);
}

// --------------------------------- head -------------------------------------

__global__ void gcn_head_kernel(const float* __restrict__ psum,
                                const float* __restrict__ pcnt,
                                const float* __restrict__ rsum,
                                const float* __restrict__ rcnt,
                                const float* __restrict__ lw,
                                const float* __restrict__ lb,
                                float* __restrict__ out) {
  int t = threadIdx.x;
  if (t >= NUM_GRAPHS * 2) return;
  int g = t >> 1, o = t & 1;
  float pc = fmaxf(pcnt[g], 1.0f);
  float rc = fmaxf(rcnt[g], 1.0f);
  float s = lb[o];
#pragma unroll 4
  for (int j = 0; j < 64; ++j) {
    s += (psum[g * 64 + j] / pc) * lw[j * 2 + o];
    s += (rsum[g * 64 + j] / rc) * lw[(64 + j) * 2 + o];
  }
  out[g * 2 + o] = s;
}

// ------------------------------- launcher -----------------------------------

extern "C" void kernel_launch(void* const* d_in, const int* in_sizes, int n_in,
                              void* d_out, int out_size, void* d_ws, size_t ws_size,
                              hipStream_t stream) {
  const float* p_x  = (const float*)d_in[0];
  const int*   p_ei = (const int*)d_in[1];
  const int*   p_bt = (const int*)d_in[2];
  const float* r_x  = (const float*)d_in[3];
  const int*   r_ei = (const int*)d_in[4];
  const int*   r_bt = (const int*)d_in[5];
  const float* p_W  = (const float*)d_in[6];
  const float* p_b  = (const float*)d_in[7];
  const float* r_W  = (const float*)d_in[8];
  const float* r_b  = (const float*)d_in[9];
  const float* lw   = (const float*)d_in[10];
  const float* lb   = (const float*)d_in[11];
  float* out = (float*)d_out;

  const int Np = in_sizes[2];
  const int Kp = in_sizes[0] / Np;   // 1280
  const int Ep = in_sizes[1] / 2;
  const int Nr = in_sizes[5];
  const int Kr = in_sizes[3] / Nr;   // 120
  const int Er = in_sizes[4] / 2;

  // carve workspace (256B-aligned bump allocator)
  char* w = (char*)d_ws;
  auto carve = [&](size_t nfloats) {
    float* p = (float*)w;
    w += ((nfloats * sizeof(float) + 255) / 256) * 256;
    return p;
  };
  float* p_xw   = carve((size_t)Np * 64);
  float* r_xw   = carve((size_t)Nr * 64);
  float* p_acc  = carve((size_t)Np * 64);
  float* r_acc  = carve((size_t)Nr * 64);
  float* p_dinv = carve((size_t)Np);
  float* r_dinv = carve((size_t)Nr);
  float* psum   = carve((size_t)NUM_GRAPHS * 64);
  float* rsum   = carve((size_t)NUM_GRAPHS * 64);
  float* pcnt   = carve((size_t)NUM_GRAPHS);
  float* rcnt   = carve((size_t)NUM_GRAPHS);
  (void)ws_size; (void)n_in; (void)out_size;

  // zero accumulators (graph-capture-safe)
  hipMemsetAsync(p_acc, 0, (size_t)Np * 64 * sizeof(float), stream);
  hipMemsetAsync(r_acc, 0, (size_t)Nr * 64 * sizeof(float), stream);
  hipMemsetAsync(psum, 0, (size_t)NUM_GRAPHS * 64 * sizeof(float), stream);
  hipMemsetAsync(rsum, 0, (size_t)NUM_GRAPHS * 64 * sizeof(float), stream);
  hipMemsetAsync(pcnt, 0, (size_t)NUM_GRAPHS * sizeof(float), stream);
  hipMemsetAsync(rcnt, 0, (size_t)NUM_GRAPHS * sizeof(float), stream);

  const int T = 256;
  // 1) degrees (self-loop = init to 1) -> dinv = rsqrt(deg)
  gcn_fill1_kernel<<<(Np + T - 1) / T, T, 0, stream>>>(p_dinv, Np);
  gcn_fill1_kernel<<<(Nr + T - 1) / T, T, 0, stream>>>(r_dinv, Nr);
  gcn_edge_deg_kernel<<<(Ep + T - 1) / T, T, 0, stream>>>(p_ei, Ep, p_dinv);
  gcn_edge_deg_kernel<<<(Er + T - 1) / T, T, 0, stream>>>(r_ei, Er, r_dinv);
  gcn_rsqrt_kernel<<<(Np + T - 1) / T, T, 0, stream>>>(p_dinv, Np);
  gcn_rsqrt_kernel<<<(Nr + T - 1) / T, T, 0, stream>>>(r_dinv, Nr);

  // 2) XW = X @ W via WMMA (wave = 32x64 tile, block = 8 waves = 256 rows)
  gcn_gemm_wmma_kernel<<<(Np + 255) / 256, T, 0, stream>>>(p_x, p_W, p_xw, Np, Kp);
  gcn_gemm_wmma_kernel<<<(Nr + 255) / 256, T, 0, stream>>>(r_x, r_W, r_xw, Nr, Kr);

  // 3) edge scatter-add (16 threads per edge, float4 each)
  {
    long long tp = (long long)Ep * 16;
    long long tr = (long long)Er * 16;
    gcn_edge_scatter_kernel<<<(unsigned)((tp + T - 1) / T), T, 0, stream>>>(
        p_ei, Ep, p_xw, p_dinv, p_acc);
    gcn_edge_scatter_kernel<<<(unsigned)((tr + T - 1) / T), T, 0, stream>>>(
        r_ei, Er, r_xw, r_dinv, r_acc);
  }

  // 4) self-loop + bias + relu + mean-pool accumulation
  {
    long long tp = (long long)Np * 16;
    long long tr = (long long)Nr * 16;
    gcn_finalize_kernel<<<(unsigned)((tp + T - 1) / T), T, 0, stream>>>(
        p_xw, p_acc, p_dinv, p_b, p_bt, psum, pcnt, Np);
    gcn_finalize_kernel<<<(unsigned)((tr + T - 1) / T), T, 0, stream>>>(
        r_xw, r_acc, r_dinv, r_b, r_bt, rsum, rcnt, Nr);
  }

  // 5) head: [128,128] @ [128,2] + b
  gcn_head_kernel<<<1, 256, 0, stream>>>(psum, pcnt, rsum, rcnt, lw, lb, out);
}
